// GraphCollaborativeFiltering_37563783971452
// MI455X (gfx1250) — matrix operand order repaired and checked by reading, not compiled, not run on hardware
//
#include <hip/hip_runtime.h>
#include <hip/hip_bf16.h>

// Graph collaborative filtering on MI455X (gfx1250, wave32).
// U=512 users, P=1024 papers, D=128, H=256, E=98304, N=1536 nodes.
//
// Roofline: after factorizing the pairwise predictor GEMM
// (combined@Wq1 == bcast(user_final@Wq1_top) + bcast(paper_final@Wq1_bot)),
// total compute is ~1 GFLOP, total traffic ~20MB (L2-resident; L2=192MB).
// GEMMs: bf16 storage, async global->LDS staging (ASYNCcnt), transposed B
// panel in LDS, 4x ds_load_b128 + v_wmma_f32_16x16x32_bf16 steady state.

#define UU      512
#define PPAP    1024
#define DD      128
#define HHID    256
#define EEDGE   98304
#define NNODES  1536   // UU + PPAP

typedef __attribute__((ext_vector_type(16))) __bf16 v16bf;
typedef __attribute__((ext_vector_type(8)))  __bf16 v8bf;
typedef __attribute__((ext_vector_type(8)))  float  v8f;

#define LDS_B_STRIDE 264   // 256 + 8 pad (elements), keeps 16B alignment

// ---------------------------------------------------------------- utilities

__global__ void fill_kernel(float* __restrict__ p, float v, int n) {
  int i = blockIdx.x * blockDim.x + threadIdx.x;
  if (i < n) p[i] = v;
}

__global__ void cvt_bf16_kernel(const float* __restrict__ in, __bf16* __restrict__ out, int n) {
  int i = blockIdx.x * blockDim.x + threadIdx.x;
  if (i < n) out[i] = (__bf16)in[i];
}

// x[0:U) = user_emb[user_ids]; x[U:N) = paper_emb[paper_ids]; bf16 out
__global__ void build_x_kernel(const int* __restrict__ uids, const int* __restrict__ pids,
                               const float* __restrict__ ue, const float* __restrict__ pe,
                               __bf16* __restrict__ x) {
  int n = blockIdx.x, f = threadIdx.x;
  if (n < UU) x[(size_t)n * DD + f] = (__bf16)ue[(size_t)uids[n] * DD + f];
  else        x[(size_t)n * DD + f] = (__bf16)pe[(size_t)pids[n - UU] * DD + f];
}

__global__ void deg_count_kernel(const int* __restrict__ dst, float* __restrict__ deg) {
  int e = blockIdx.x * blockDim.x + threadIdx.x;
  if (e < EEDGE) atomicAdd(&deg[dst[e]], 1.0f);
}

__global__ void dis_kernel(const float* __restrict__ deg, float* __restrict__ dis) {
  int n = blockIdx.x * blockDim.x + threadIdx.x;
  if (n < NNODES) dis[n] = rsqrtf(deg[n]);
}

// ------------------------------------------------------- WMMA bf16 GEMM v3
// C[M,N] = A[M,K] @ B[K,N] (+bias)(+relu), A/B bf16, C f32 or bf16.
// Block = 256 threads (8 waves), macro-tile 128 rows x 16 cols.
// A panel ([128,K] bf16) staged global->LDS via async-to-LDS b128 copies;
// B panel ([K,16]) staged transposed into LDS ([16][K], +8 pad).
// Steady state per wave / K-step: 4x ds_load_b128 + 1x v_wmma (no VMEM).
// M%128==0, N%16==0, K%32==0, K<=256. EXEC all-ones at every WMMA.
__global__ void __launch_bounds__(256)
gemm_wmma_bf16(const __bf16* __restrict__ A, const __bf16* __restrict__ B,
               const float* __restrict__ bias, void* __restrict__ Cout,
               int M, int N, int K, int add_bias, int do_relu, int out_bf16) {
  __shared__ alignas(16) __bf16 lA[128 * 256];           // 64 KB max
  __shared__ alignas(16) __bf16 lB[16 * LDS_B_STRIDE];   // 8.25 KB

  const int tid  = threadIdx.x;
  const int wave = tid >> 5;
  const int lane = tid & 31;
  const int hl   = lane >> 4;      // lane half
  const int l16  = lane & 15;
  const int tm0  = blockIdx.y << 7;  // 128-row macro tile
  const int tn0  = blockIdx.x << 4;  // 16-col tile

  // ---- stage A panel [128][K] via async global->LDS (16B per lane-copy)
  {
    const __bf16* Ab = A + (size_t)tm0 * K;
    const unsigned lA_base = (unsigned)(uintptr_t)&lA[0];
    const int chunks = (128 * K) >> 3;            // 8 bf16 per 16B chunk
    for (int c = tid; c < chunks; c += 256) {     // uniform trip count
      unsigned loff = lA_base + ((unsigned)c << 4);
      const __bf16* g = Ab + ((size_t)c << 3);
      asm volatile("global_load_async_to_lds_b128 %0, %1, off"
                   :: "v"(loff), "v"(g) : "memory");
    }
  }
  // ---- stage B panel transposed: lB[n][k] = B[k][tn0+n]   (overlaps async)
  for (int i = tid; i < K * 16; i += 256) {
    int k = i >> 4, n = i & 15;
    lB[n * LDS_B_STRIDE + k] = B[(size_t)k * N + tn0 + n];
  }
  asm volatile("s_wait_asynccnt 0x0" ::: "memory");
  __syncthreads();

  // ---- MAC loop: fragments straight out of LDS (all 16B-aligned)
  v8f acc = {};
  const int aRow = (wave * 16 + l16) * K;
  const int bRow = l16 * LDS_B_STRIDE;
  for (int kb = 0; kb < K; kb += 32) {
    // A 16x32: per lane two contiguous 8-elem chunks: K=hl*8.., K=16+hl*8..
    v8bf a0 = *(const v8bf*)&lA[aRow + kb + hl * 8];
    v8bf a1 = *(const v8bf*)&lA[aRow + kb + 16 + hl * 8];
    // B 32x16: per lane two contiguous 8-elem chunks: K=hl*8.., K=16+hl*8..
    v8bf b0 = *(const v8bf*)&lB[bRow + kb + hl * 8];
    v8bf b1 = *(const v8bf*)&lB[bRow + kb + 16 + hl * 8];
    v16bf a = __builtin_shufflevector(a0, a1, 0,1,2,3,4,5,6,7,8,9,10,11,12,13,14,15);
    v16bf b = __builtin_shufflevector(b0, b1, 0,1,2,3,4,5,6,7,8,9,10,11,12,13,14,15);
    acc = __builtin_amdgcn_wmma_f32_16x16x32_bf16(
        /*neg_a=*/false, a, /*neg_b=*/false, b,
        /*c_mod=*/(short)0, acc, /*reuse_a=*/false, /*reuse_b=*/false);
  }

  // ---- epilogue: C/D layout row = 8*half + r, col = lane%16
  const int col = tn0 + l16;
  const float bv = add_bias ? bias[col] : 0.0f;
#pragma unroll
  for (int r = 0; r < 8; ++r) {
    int row = tm0 + wave * 16 + hl * 8 + r;
    float v = acc[r] + bv;
    if (do_relu) v = fmaxf(v, 0.0f);
    if (out_bf16) ((__bf16*)Cout)[(size_t)row * N + col] = (__bf16)v;
    else          ((float*)Cout)[(size_t)row * N + col] = v;
  }
}

// -------------------------------------------------- GCN edge aggregation
__global__ void edge_agg_kernel(const int* __restrict__ src, const int* __restrict__ dst,
                                const float* __restrict__ dis, const float* __restrict__ h,
                                float* __restrict__ agg) {
  int e = blockIdx.x, f = threadIdx.x;
  int s = src[e], d = dst[e];
  float nrm = dis[s] * dis[d];
  atomicAdd(&agg[(size_t)d * HHID + f], h[(size_t)s * HHID + f] * nrm);
}

// x_out = relu(agg + h*dis^2 + bias), written as bf16 for the next GEMM
__global__ void finalize_kernel(const float* __restrict__ agg, const float* __restrict__ h,
                                const float* __restrict__ dis, const float* __restrict__ b,
                                __bf16* __restrict__ xo) {
  int n = blockIdx.x, f = threadIdx.x;
  float dn = dis[n];
  float v = agg[(size_t)n * HHID + f] + h[(size_t)n * HHID + f] * dn * dn + b[f];
  xo[(size_t)n * HHID + f] = (__bf16)fmaxf(v, 0.0f);
}

// ------------------------------------------------------ pairwise predictor
// pred[u,p] = sigmoid( sum_k relu(hu[u,k]+hp[p,k]+bq1[k]) * wq2[k] + bq2 )
__global__ void predict_kernel(const float* __restrict__ hu, const float* __restrict__ hp,
                               const float* __restrict__ bq1, const float* __restrict__ wq2,
                               const float* __restrict__ bq2, float* __restrict__ out) {
  int p = blockIdx.x * blockDim.x + threadIdx.x;
  int u = blockIdx.y;
  const float* hur = hu + (size_t)u * HHID;
  const float* hpr = hp + (size_t)p * HHID;
  float acc = 0.0f;
#pragma unroll 8
  for (int k = 0; k < HHID; ++k) {
    float v = fmaxf(hur[k] + hpr[k] + bq1[k], 0.0f);
    acc = fmaf(v, wq2[k], acc);
  }
  float z = acc + bq2[0];
  out[(size_t)u * PPAP + p] = 1.0f / (1.0f + __expf(-z));
}

// ------------------------------------------------------------------ launch

extern "C" void kernel_launch(void* const* d_in, const int* in_sizes, int n_in,
                              void* d_out, int out_size, void* d_ws, size_t ws_size,
                              hipStream_t stream) {
  (void)in_sizes; (void)n_in; (void)out_size; (void)ws_size;

  const int*   uids = (const int*)d_in[0];
  const int*   pids = (const int*)d_in[1];
  const int*   edge = (const int*)d_in[2];     // [2,E] row-major
  // d_in[3] (user_paper_interactions) unused by the reference
  const float* ue   = (const float*)d_in[4];
  const float* pe   = (const float*)d_in[5];
  const float* W0   = (const float*)d_in[6];   const float* b0  = (const float*)d_in[7];
  const float* W1   = (const float*)d_in[8];   const float* b1  = (const float*)d_in[9];
  const float* W2   = (const float*)d_in[10];  const float* b2  = (const float*)d_in[11];
  const float* Wu   = (const float*)d_in[12];  const float* bu  = (const float*)d_in[13];
  const float* Wp   = (const float*)d_in[14];  const float* bp  = (const float*)d_in[15];
  const float* Wq1  = (const float*)d_in[16];  const float* bq1 = (const float*)d_in[17];
  const float* Wq2  = (const float*)d_in[18];  const float* bq2 = (const float*)d_in[19];
  const int* srcv = edge;
  const int* dstv = edge + EEDGE;

  // ---- workspace layout: f32 region first, then 16B-aligned bf16 region
  float* wsf = (float*)d_ws;
  float* h    = wsf;                                 // [1536,256] f32
  float* agg  = h   + (size_t)NNODES * HHID;         // [1536,256] f32
  float* deg  = agg + (size_t)NNODES * HHID;         // [1536]
  float* dis  = deg + NNODES;                        // [1536]
  float* hu   = dis + NNODES;                        // [512,256]  f32
  float* hp   = hu  + (size_t)UU * HHID;             // [1024,256] f32
  __bf16* wsb = (__bf16*)(hp + (size_t)PPAP * HHID); // 16B-aligned
  __bf16* xA  = wsb;                                 // [1536,256] bf16
  __bf16* xB  = xA  + (size_t)NNODES * HHID;
  __bf16* uf  = xB  + (size_t)NNODES * HHID;         // [512,128]
  __bf16* pf  = uf  + (size_t)UU * DD;               // [1024,128]
  __bf16* W0b = pf  + (size_t)PPAP * DD;             // [128,256]
  __bf16* W1b = W0b + (size_t)DD * HHID;             // [256,256]
  __bf16* W2b = W1b + (size_t)HHID * HHID;           // [256,256]
  __bf16* Wub = W2b + (size_t)HHID * HHID;           // [256,128]
  __bf16* Wpb = Wub + (size_t)HHID * DD;             // [256,128]
  __bf16* Wq1b= Wpb + (size_t)HHID * DD;             // [256,256]

  const dim3 blk(256);

  // ---- one-time bf16 conversion of all GEMM weights
  cvt_bf16_kernel<<<(DD*HHID+255)/256,   blk, 0, stream>>>(W0,  W0b,  DD*HHID);
  cvt_bf16_kernel<<<(HHID*HHID+255)/256, blk, 0, stream>>>(W1,  W1b,  HHID*HHID);
  cvt_bf16_kernel<<<(HHID*HHID+255)/256, blk, 0, stream>>>(W2,  W2b,  HHID*HHID);
  cvt_bf16_kernel<<<(HHID*DD+255)/256,   blk, 0, stream>>>(Wu,  Wub,  HHID*DD);
  cvt_bf16_kernel<<<(HHID*DD+255)/256,   blk, 0, stream>>>(Wp,  Wpb,  HHID*DD);
  cvt_bf16_kernel<<<(HHID*HHID+255)/256, blk, 0, stream>>>(Wq1, Wq1b, HHID*HHID);

  // ---- node features + degree normalization
  build_x_kernel<<<NNODES, DD, 0, stream>>>(uids, pids, ue, pe, xA);
  fill_kernel<<<(NNODES+255)/256, blk, 0, stream>>>(deg, 1.0f, NNODES);
  deg_count_kernel<<<(EEDGE+255)/256, blk, 0, stream>>>(dstv, deg);
  dis_kernel<<<(NNODES+255)/256, blk, 0, stream>>>(deg, dis);

  // ---- GCN layer 0 -> xB
  gemm_wmma_bf16<<<dim3(HHID/16, NNODES/128), blk, 0, stream>>>(
      xA, W0b, nullptr, h, NNODES, HHID, DD, 0, 0, 0);
  fill_kernel<<<(NNODES*HHID+255)/256, blk, 0, stream>>>(agg, 0.0f, NNODES*HHID);
  edge_agg_kernel<<<EEDGE, HHID, 0, stream>>>(srcv, dstv, dis, h, agg);
  finalize_kernel<<<NNODES, HHID, 0, stream>>>(agg, h, dis, b0, xB);

  // ---- GCN layer 1 -> xA
  gemm_wmma_bf16<<<dim3(HHID/16, NNODES/128), blk, 0, stream>>>(
      xB, W1b, nullptr, h, NNODES, HHID, HHID, 0, 0, 0);
  fill_kernel<<<(NNODES*HHID+255)/256, blk, 0, stream>>>(agg, 0.0f, NNODES*HHID);
  edge_agg_kernel<<<EEDGE, HHID, 0, stream>>>(srcv, dstv, dis, h, agg);
  finalize_kernel<<<NNODES, HHID, 0, stream>>>(agg, h, dis, b1, xA);

  // ---- GCN layer 2 -> xB
  gemm_wmma_bf16<<<dim3(HHID/16, NNODES/128), blk, 0, stream>>>(
      xA, W2b, nullptr, h, NNODES, HHID, HHID, 0, 0, 0);
  fill_kernel<<<(NNODES*HHID+255)/256, blk, 0, stream>>>(agg, 0.0f, NNODES*HHID);
  edge_agg_kernel<<<EEDGE, HHID, 0, stream>>>(srcv, dstv, dis, h, agg);
  finalize_kernel<<<NNODES, HHID, 0, stream>>>(agg, h, dis, b2, xB);

  // ---- output heads (bias, no relu), bf16 outputs feed next GEMMs
  gemm_wmma_bf16<<<dim3(DD/16, UU/128), blk, 0, stream>>>(
      xB, Wub, bu, uf, UU, DD, HHID, 1, 0, 1);
  gemm_wmma_bf16<<<dim3(DD/16, PPAP/128), blk, 0, stream>>>(
      xB + (size_t)UU * HHID, Wpb, bp, pf, PPAP, DD, HHID, 1, 0, 1);

  // ---- factorized predictor first layer (f32 outputs)
  gemm_wmma_bf16<<<dim3(HHID/16, UU/128), blk, 0, stream>>>(
      uf, Wq1b, nullptr, hu, UU, HHID, DD, 0, 0, 0);
  gemm_wmma_bf16<<<dim3(HHID/16, PPAP/128), blk, 0, stream>>>(
      pf, Wq1b + (size_t)DD * HHID, nullptr, hp, PPAP, HHID, DD, 0, 0, 0);

  // ---- pairwise relu-dot + sigmoid -> d_out [512,1024]
  predict_kernel<<<dim3(PPAP/256, UU), blk, 0, stream>>>(
      hu, hp, bq1, Wq2, bq2, (float*)d_out);
}
